// Feature_Viewpooling_33732673143357
// MI455X (gfx1250) — matrix operand order repaired and checked by reading, not compiled, not run on hardware
//
#include <hip/hip_runtime.h>
#include <hip/hip_bf16.h>

typedef __attribute__((ext_vector_type(16))) __bf16 v16bf;
typedef __attribute__((ext_vector_type(8)))  __bf16 v8bf;
typedef __attribute__((ext_vector_type(8)))  float  v8f;

#define BATCH 96
#define NPTS  20
#define DIM   2048
#define KNN   10
#define BN_EPS 1e-5f

// ---------------------------------------------------------------------------
// Fragment loaders: 16 elements = 8 from p0, 8 from p1 (contiguous chunks).
// ---------------------------------------------------------------------------
__device__ inline v16bf ld8pair(const float* p0, const float* p1) {
    float4 a0 = ((const float4*)p0)[0], a1 = ((const float4*)p0)[1];
    float4 b0 = ((const float4*)p1)[0], b1 = ((const float4*)p1)[1];
    v16bf r;
    r[0]  = (__bf16)a0.x; r[1]  = (__bf16)a0.y; r[2]  = (__bf16)a0.z; r[3]  = (__bf16)a0.w;
    r[4]  = (__bf16)a1.x; r[5]  = (__bf16)a1.y; r[6]  = (__bf16)a1.z; r[7]  = (__bf16)a1.w;
    r[8]  = (__bf16)b0.x; r[9]  = (__bf16)b0.y; r[10] = (__bf16)b0.z; r[11] = (__bf16)b0.w;
    r[12] = (__bf16)b1.x; r[13] = (__bf16)b1.y; r[14] = (__bf16)b1.z; r[15] = (__bf16)b1.w;
    return r;
}
__device__ inline v16bf ld8pair(const __bf16* p0, const __bf16* p1) {
    v8bf lo = *(const v8bf*)p0;
    v8bf hi = *(const v8bf*)p1;
    return __builtin_shufflevector(lo, hi, 0,1,2,3,4,5,6,7,8,9,10,11,12,13,14,15);
}

// A+B fragment set for one K-step (5 x 8 VGPRs)
template <typename ET>
struct Frags { v16bf bm, a0, a1, a2, a3; };

template <typename ET>
__device__ inline Frags<ET> load_frags(const ET* __restrict__ xcol,
                                       const ET* __restrict__ wrow0,
                                       int k0, int h) {
    const size_t wsub = (size_t)16 * (2 * DIM);     // 16 W-rows per sub-tile
    Frags<ET> f;
    // B fragment: elements 0..7 <- k0 + h*8, elements 8..15 <- k0 + 16 + h*8
    const ET* bp = xcol + k0 + h * 8;
    f.bm = ld8pair(bp, bp + 16);
    // A fragments: elements 0..7 <- k0 + h*8, elements 8..15 <- k0 + 16 + h*8
    const int ka0 = k0 + h * 8;
    const int ka1 = k0 + 16 + h * 8;
    f.a0 = ld8pair(wrow0 + 0 * wsub + ka0, wrow0 + 0 * wsub + ka1);
    f.a1 = ld8pair(wrow0 + 1 * wsub + ka0, wrow0 + 1 * wsub + ka1);
    f.a2 = ld8pair(wrow0 + 2 * wsub + ka0, wrow0 + 2 * wsub + ka1);
    f.a3 = ld8pair(wrow0 + 3 * wsub + ka0, wrow0 + 3 * wsub + ka1);
    return f;
}

// ---------------------------------------------------------------------------
// fp32 -> bf16 bulk conversion (8 elements per thread, b128 store).
// ---------------------------------------------------------------------------
__global__ __launch_bounds__(256) void cvt_bf16_kernel(const float* __restrict__ in,
                                                       __bf16* __restrict__ out,
                                                       int nvec8) {
    const int i = blockIdx.x * 256 + threadIdx.x;
    if (i < nvec8) {
        const float4* p = (const float4*)in + 2 * (size_t)i;
        float4 a = p[0], b = p[1];
        v8bf o;
        o[0] = (__bf16)a.x; o[1] = (__bf16)a.y; o[2] = (__bf16)a.z; o[3] = (__bf16)a.w;
        o[4] = (__bf16)b.x; o[5] = (__bf16)b.y; o[6] = (__bf16)b.z; o[7] = (__bf16)b.w;
        ((v8bf*)out)[i] = o;
    }
}

// ---------------------------------------------------------------------------
// Kernel 1: pairwise distances + top-k neighbor indices, one block per batch.
// ---------------------------------------------------------------------------
__global__ __launch_bounds__(256) void knn_kernel(const float* __restrict__ x,
                                                  int* __restrict__ nn_idx) {
    const int b   = blockIdx.x;
    const int tid = threadIdx.x;

    __shared__ float chunk[NPTS][257];
    __shared__ float inner[NPTS * NPTS];

    const int p0 = tid;
    const int p1 = tid + 256;
    const int n0 = p0 / NPTS, m0 = p0 - n0 * NPTS;
    const int n1 = p1 / NPTS, m1 = p1 - n1 * NPTS;
    float acc0 = 0.f, acc1 = 0.f;

    const float* xb = x + (size_t)b * NPTS * DIM;

    for (int k0 = 0; k0 < DIM; k0 += 256) {
        #pragma unroll
        for (int r = 0; r < NPTS; ++r)
            chunk[r][tid] = xb[r * DIM + k0 + tid];
        __syncthreads();

        if (p0 < NPTS * NPTS) {
            #pragma unroll 8
            for (int kk = 0; kk < 256; ++kk)
                acc0 += chunk[n0][kk] * chunk[m0][kk];
        }
        if (p1 < NPTS * NPTS) {
            #pragma unroll 8
            for (int kk = 0; kk < 256; ++kk)
                acc1 += chunk[n1][kk] * chunk[m1][kk];
        }
        __syncthreads();
    }

    if (p0 < NPTS * NPTS) inner[p0] = acc0;
    if (p1 < NPTS * NPTS) inner[p1] = acc1;
    __syncthreads();

    if (tid < NPTS) {
        const int n = tid;
        float adj[NPTS];
        bool  used[NPTS];
        const float sqn = inner[n * NPTS + n];
        #pragma unroll
        for (int m = 0; m < NPTS; ++m) {
            adj[m]  = sqn + inner[m * NPTS + m] - 2.f * inner[n * NPTS + m];
            used[m] = false;
        }
        for (int j = 0; j < KNN; ++j) {
            int best = 0; float bv = 3.0e38f;
            #pragma unroll
            for (int m = 0; m < NPTS; ++m) {
                if (!used[m] && adj[m] < bv) { bv = adj[m]; best = m; }
            }
            used[best] = true;
            nn_idx[(b * NPTS + n) * KNN + j] = best;
        }
    }
}

// ---------------------------------------------------------------------------
// Kernel 2: fused split-GEMM (u = W1*X, v = W2*X via bf16 WMMA) + gather +
// BN + ReLU + max-pool. Grid: (32 tiles of 64 channels, 96 batches),
// 4 waves/block. Software-pipelined K-loop: next K-step's 10 b128 loads are
// issued before current K-step's 4 WMMAs consume their fragments.
// ---------------------------------------------------------------------------
template <typename ET>
__global__ __launch_bounds__(128) void edgeconv_kernel(
    const ET* __restrict__ x,        const ET* __restrict__ w,
    const float* __restrict__ bias,  const float* __restrict__ gamma,
    const float* __restrict__ beta,  const float* __restrict__ mean,
    const float* __restrict__ var,   const int* __restrict__ nn_idx,
    float* __restrict__ out) {

    const int o_tile = blockIdx.x;       // 0..31 (64 channels each)
    const int b      = blockIdx.y;       // 0..95
    const int tid    = threadIdx.x;
    const int lane   = tid & 31;
    const int wid    = tid >> 5;
    const int g      = wid >> 1;         // 0 -> W1 (u), 1 -> W2 (v)
    const int t      = wid & 1;          // column half
    const int h      = lane >> 4;        // lane half
    const int l16    = lane & 15;

    const int o0       = o_tile * 64;
    const int colStore = t * 16 + l16;                               // 0..31
    const int colLoad  = (colStore < NPTS) ? colStore : (NPTS - 1);  // clamp: junk cols never read

    const ET* xcol  = x + ((size_t)b * NPTS + colLoad) * DIM;
    const ET* wrow0 = w + (size_t)(o0 + l16) * (2 * DIM) + (size_t)g * DIM;

    v8f acc0 = {}, acc1 = {}, acc2 = {}, acc3 = {};

    // -------- software-pipelined K loop --------
    Frags<ET> cur = load_frags(xcol, wrow0, 0, h);
    for (int k0 = 0; k0 < DIM - 32; k0 += 32) {
        Frags<ET> nxt = load_frags(xcol, wrow0, k0 + 32, h);   // prefetch
        acc0 = __builtin_amdgcn_wmma_f32_16x16x32_bf16(false, cur.a0, false, cur.bm, (short)0, acc0, false, false);
        acc1 = __builtin_amdgcn_wmma_f32_16x16x32_bf16(false, cur.a1, false, cur.bm, (short)0, acc1, false, false);
        acc2 = __builtin_amdgcn_wmma_f32_16x16x32_bf16(false, cur.a2, false, cur.bm, (short)0, acc2, false, false);
        acc3 = __builtin_amdgcn_wmma_f32_16x16x32_bf16(false, cur.a3, false, cur.bm, (short)0, acc3, false, false);
        cur = nxt;
    }
    acc0 = __builtin_amdgcn_wmma_f32_16x16x32_bf16(false, cur.a0, false, cur.bm, (short)0, acc0, false, false);
    acc1 = __builtin_amdgcn_wmma_f32_16x16x32_bf16(false, cur.a1, false, cur.bm, (short)0, acc1, false, false);
    acc2 = __builtin_amdgcn_wmma_f32_16x16x32_bf16(false, cur.a2, false, cur.bm, (short)0, acc2, false, false);
    acc3 = __builtin_amdgcn_wmma_f32_16x16x32_bf16(false, cur.a3, false, cur.bm, (short)0, acc3, false, false);

    // ---- stage u/v tiles in LDS ----
    __shared__ float sh_u[64][33];
    __shared__ float sh_v[64][33];
    __shared__ float sh_inv[64], sh_shift[64], sh_bias[64];
    __shared__ int   sh_idx[NPTS * KNN];
    __shared__ int   sh_max[64];

    {
        float* shd = g ? &sh_v[0][0] : &sh_u[0][0];
        #pragma unroll
        for (int r = 0; r < 8; ++r) {
            // C/D layout: VGPR r -> row r + 8*h within sub-tile
            shd[(0 * 16 + r + 8 * h) * 33 + colStore] = acc0[r];
            shd[(1 * 16 + r + 8 * h) * 33 + colStore] = acc1[r];
            shd[(2 * 16 + r + 8 * h) * 33 + colStore] = acc2[r];
            shd[(3 * 16 + r + 8 * h) * 33 + colStore] = acc3[r];
        }
    }
    if (tid < 64) {
        const int o   = o0 + tid;
        const float iv = gamma[o] * rsqrtf(var[o] + BN_EPS);
        sh_inv[tid]   = iv;
        sh_shift[tid] = beta[o] - mean[o] * iv;
        sh_bias[tid]  = bias[o];
        sh_max[tid]   = 0;   // bits of 0.0f; ReLU output >= 0 so exact
    }
    for (int i = tid; i < NPTS * KNN; i += 128)
        sh_idx[i] = nn_idx[b * NPTS * KNN + i];
    __syncthreads();

    // ---- combine: y = relu((u[n] + v[idx] - v[n] + bias)*inv + shift); max ----
    for (int p = tid; p < 64 * NPTS; p += 128) {
        const int m = p & 63;
        const int n = p >> 6;
        const float base = sh_u[m][n] - sh_v[m][n] + sh_bias[m];
        const float iv = sh_inv[m], sf = sh_shift[m];
        float lmax = 0.f;
        #pragma unroll
        for (int j = 0; j < KNN; ++j) {
            const int idx = sh_idx[n * KNN + j];
            float val = (base + sh_v[m][idx]) * iv + sf;
            lmax = fmaxf(lmax, fmaxf(val, 0.f));
        }
        atomicMax(&sh_max[m], __float_as_int(lmax));
    }
    __syncthreads();

    if (tid < 64)
        out[(size_t)b * DIM + o0 + tid] = __int_as_float(sh_max[tid]);
}

// ---------------------------------------------------------------------------
extern "C" void kernel_launch(void* const* d_in, const int* in_sizes, int n_in,
                              void* d_out, int out_size, void* d_ws, size_t ws_size,
                              hipStream_t stream) {
    const float* x      = (const float*)d_in[0];
    const float* conv_w = (const float*)d_in[1];
    const float* conv_b = (const float*)d_in[2];
    const float* gamma  = (const float*)d_in[3];
    const float* beta   = (const float*)d_in[4];
    const float* mean   = (const float*)d_in[5];
    const float* var    = (const float*)d_in[6];

    int* nn_idx = (int*)d_ws;                          // 76.8 KB
    const size_t IDX_BYTES = 128 * 1024;               // aligned region for indices
    const size_t W_ELEMS = (size_t)DIM * 2 * DIM;      // 8.4M
    const size_t X_ELEMS = (size_t)BATCH * NPTS * DIM; // 3.9M
    const size_t NEED = IDX_BYTES + (W_ELEMS + X_ELEMS) * sizeof(__bf16);

    knn_kernel<<<dim3(BATCH), dim3(256), 0, stream>>>(x, nn_idx);

    if (ws_size >= NEED) {
        // Pre-convert W and x to bf16 once: halves hot-loop load bytes,
        // removes all cvt VALU from the 64-step WMMA loop.
        __bf16* wbf = (__bf16*)((char*)d_ws + IDX_BYTES);
        __bf16* xbf = wbf + W_ELEMS;
        const int wv = (int)(W_ELEMS / 8), xv = (int)(X_ELEMS / 8);
        cvt_bf16_kernel<<<dim3((wv + 255) / 256), dim3(256), 0, stream>>>(conv_w, wbf, wv);
        cvt_bf16_kernel<<<dim3((xv + 255) / 256), dim3(256), 0, stream>>>(x, xbf, xv);

        edgeconv_kernel<__bf16><<<dim3(DIM / 64, BATCH), dim3(128), 0, stream>>>(
            xbf, wbf, conv_b, gamma, beta, mean, var, nn_idx, (float*)d_out);
    } else {
        // Fallback: convert fp32 -> bf16 on the fly inside the GEMM loop.
        edgeconv_kernel<float><<<dim3(DIM / 64, BATCH), dim3(128), 0, stream>>>(
            x, conv_w, conv_b, gamma, beta, mean, var, nn_idx, (float*)d_out);
    }
}
